// EpipolarCrossAttention_86071144612592
// MI455X (gfx1250) — matrix operand order repaired and checked by reading, not compiled, not run on hardware
//
#include <hip/hip_runtime.h>
#include <hip/hip_bf16.h>

// ---------------- problem constants (from setup_inputs) ----------------
#define BB     2
#define SS     4
#define HH     24
#define WW     24
#define HW     (HH*WW)          // 576
#define CIN    512
#define NHEAD  8
#define DK     64
#define DV     32
#define NS     32
#define NTOT   (3*NS)           // 96 samples per ref pixel
#define CQKV   1280             // 512(Q) + 512(K) + 256(V)
#define NROW   (BB*SS*HW)       // 4608
#define OC     512
#define KMAX   512              // largest GEMM K dimension

typedef __attribute__((ext_vector_type(16))) __bf16 v16bf;
typedef __attribute__((ext_vector_type(8)))  float  v8f;

// ---------------- small conversion / copy kernels ----------------
__global__ void f32_to_bf16_kernel(const float* __restrict__ src,
                                   __bf16* __restrict__ dst, int n) {
  int i = blockIdx.x * blockDim.x + threadIdx.x;
  if (i < n) dst[i] = (__bf16)src[i];
}

__global__ void f32_copy_kernel(const float* __restrict__ src,
                                float* __restrict__ dst, int n) {
  int i = blockIdx.x * blockDim.x + threadIdx.x;
  if (i < n) dst[i] = src[i];
}

// ---------------- fundamental matrices ----------------
__device__ inline void inv3(const float* m, float* o) {
  float a = m[0], b = m[1], c = m[2];
  float d = m[3], e = m[4], f = m[5];
  float g = m[6], h = m[7], i = m[8];
  float A = e*i - f*h, Bc = -(d*i - f*g), Cc = d*h - e*g;
  float det = a*A + b*Bc + c*Cc;
  float id = 1.0f / det;
  o[0] = A*id;            o[1] = (c*h - b*i)*id;  o[2] = (b*f - c*e)*id;
  o[3] = Bc*id;           o[4] = (a*i - c*g)*id;  o[5] = (c*d - a*f)*id;
  o[6] = Cc*id;           o[7] = (b*g - a*h)*id;  o[8] = (a*e - b*d)*id;
}

__global__ void fmat_kernel(const float* __restrict__ intr,  // (B,S,3,3)
                            const float* __restrict__ ext,   // (B,S,4,4)
                            float* __restrict__ Fout) {      // (B,S,S,9)
  int t = threadIdx.x;
  if (t >= BB*SS*SS) return;
  int src = t % SS;
  int ref = (t / SS) % SS;
  int b   = t / (SS*SS);

  const float* E1 = ext + (size_t)(b*SS + ref)*16;
  const float* E2 = ext + (size_t)(b*SS + src)*16;
  float R1[9], R2[9], t1[3], t2[3];
  #pragma unroll
  for (int i = 0; i < 3; ++i) {
    #pragma unroll
    for (int j = 0; j < 3; ++j) { R1[i*3+j] = E1[i*4+j]; R2[i*3+j] = E2[i*4+j]; }
    t1[i] = E1[i*4+3]; t2[i] = E2[i*4+3];
  }
  float R[9], tt[3];
  #pragma unroll
  for (int i = 0; i < 3; ++i)
    #pragma unroll
    for (int j = 0; j < 3; ++j)
      R[i*3+j] = R2[i*3+0]*R1[j*3+0] + R2[i*3+1]*R1[j*3+1] + R2[i*3+2]*R1[j*3+2];
  #pragma unroll
  for (int i = 0; i < 3; ++i)
    tt[i] = t2[i] - (R[i*3+0]*t1[0] + R[i*3+1]*t1[1] + R[i*3+2]*t1[2]);
  float Em[9];
  #pragma unroll
  for (int j = 0; j < 3; ++j) {
    Em[0*3+j] = -tt[2]*R[1*3+j] + tt[1]*R[2*3+j];
    Em[1*3+j] =  tt[2]*R[0*3+j] - tt[0]*R[2*3+j];
    Em[2*3+j] = -tt[1]*R[0*3+j] + tt[0]*R[1*3+j];
  }
  float K1i[9], K2i[9];
  inv3(intr + (size_t)(b*SS + ref)*9, K1i);
  inv3(intr + (size_t)(b*SS + src)*9, K2i);
  float T[9];   // T = K2i^T * E
  #pragma unroll
  for (int i = 0; i < 3; ++i)
    #pragma unroll
    for (int k = 0; k < 3; ++k)
      T[i*3+k] = K2i[0*3+i]*Em[0*3+k] + K2i[1*3+i]*Em[1*3+k] + K2i[2*3+i]*Em[2*3+k];
  float* F = Fout + (size_t)t * 9;
  #pragma unroll
  for (int i = 0; i < 3; ++i)
    #pragma unroll
    for (int l = 0; l < 3; ++l)
      F[i*3+l] = T[i*3+0]*K1i[0*3+l] + T[i*3+1]*K1i[1*3+l] + T[i*3+2]*K1i[2*3+l];
}

// ---------------- bf16 WMMA GEMM:  out = A(MxK) * W(NxK)^T + bias ----------------
// Block = 256 threads = 8 waves. Block tile = 128 rows x 32 cols.
// Weight panel (32 x K) staged once through LDS (shared by all 8 waves);
// each wave computes a 16x32 tile = two v_wmma_f32_16x16x32_bf16 per K-step
// sharing one A fragment in registers.
__global__ __launch_bounds__(256) void gemm_bf16_kernel(
    const __bf16* __restrict__ A,     // M x K row-major
    const __bf16* __restrict__ Wt,    // N x K row-major
    const float*  __restrict__ bias,  // len N (may be null)
    float*        __restrict__ outF,  // M x N (may be null)
    __bf16*       __restrict__ outB,  // M x N (may be null)
    int M, int N, int K)
{
  __shared__ __bf16 Bs[32 * KMAX];    // 32 KB

  int wave = threadIdx.x >> 5;
  int lane = threadIdx.x & 31;
  int n0 = blockIdx.x * 32;
  int m0 = (blockIdx.y * 8 + wave) * 16;

  // ---- stage weight panel Wt[n0 .. n0+31][0..K) into LDS (b128 chunks) ----
  {
    int vecPerRow = K >> 3;                 // 8 bf16 per uint4
    int total = 32 * vecPerRow;
    for (int id = threadIdx.x; id < total; id += 256) {
      int r = id / vecPerRow, c = id - r * vecPerRow;
      const uint4* src = (const uint4*)(Wt + (size_t)(n0 + r) * K) + c;
      *((uint4*)(Bs + r * K) + c) = *src;
    }
  }
  __syncthreads();

  int row = lane & 15;                 // M-row / N-col within tile
  int hb  = (lane & 16) ? 8  : 0;      // A-fragment K half-base
  int khb = (lane & 16) ? 16 : 0;      // unused; B fragment uses hb too

  const __bf16* Arow = A + (size_t)(m0 + row) * K;
  const __bf16* B0   = Bs + row * K;          // cols n0 .. n0+15
  const __bf16* B1   = Bs + (16 + row) * K;   // cols n0+16 .. n0+31

  v8f c0 = {}, c1 = {};
  for (int k0 = 0; k0 < K; k0 += 32) {
    v16bf a, b0, b1;
    #pragma unroll
    for (int e = 0; e < 16; ++e) {
      int j = e >> 1;
      int k = ((j < 4) ? (hb + 2*j) : (16 + hb + 2*(j - 4))) + (e & 1);
      a[e]  = Arow[k0 + k];
      b0[e] = B0[k0 + k];
      b1[e] = B1[k0 + k];
    }
    c0 = __builtin_amdgcn_wmma_f32_16x16x32_bf16(
             false, a, false, b0, (short)0, c0, false, false);
    c1 = __builtin_amdgcn_wmma_f32_16x16x32_bf16(
             false, a, false, b1, (short)0, c1, false, false);
  }

  int ni0 = n0 + row, ni1 = n0 + 16 + row;
  float bv0 = bias ? bias[ni0] : 0.0f;
  float bv1 = bias ? bias[ni1] : 0.0f;
  #pragma unroll
  for (int r = 0; r < 8; ++r) {
    int m = m0 + r + hb;
    float v0 = c0[r] + bv0;
    float v1 = c1[r] + bv1;
    if (outF) { outF[(size_t)m * N + ni0] = v0; outF[(size_t)m * N + ni1] = v1; }
    if (outB) { outB[(size_t)m * N + ni0] = (__bf16)v0; outB[(size_t)m * N + ni1] = (__bf16)v1; }
  }
}

// ---------------- epipolar attention core ----------------
// One block per (b, ref, pixel). 256 threads = 8 waves (one wave per head).
__global__ __launch_bounds__(256) void attn_kernel(
    const float* __restrict__ QKV,   // (NROW, CQKV) channel-last
    const float* __restrict__ Fmat,  // (B,S,S,9)
    __bf16*      __restrict__ obf)   // (NROW, NHEAD*DV)
{
  __shared__ float sx[NTOT], sy[NTOT];
  __shared__ float slog[NHEAD * NTOT];

  int blk  = blockIdx.x;               // (b*S+ref)*HW + pix
  int pix  = blk % HW;
  int bref = blk / HW;
  int ref  = bref % SS;
  int b    = bref / SS;
  float px = (float)(pix % WW);
  float py = (float)(pix / WW);
  int t    = threadIdx.x;
  int wave = t >> 5;
  int lane = t & 31;

  // ---- stage 1: 96 sample coordinates (pixel space; the reference's
  //      grid normalization round-trips exactly) ----
  if (t < NTOT) {
    int j   = t / NS;
    int n   = t % NS;
    int src = j + (j >= ref ? 1 : 0);
    const float* F = Fmat + ((size_t)(b*SS + ref)*SS + src) * 9;
    float aa = F[0]*px + F[1]*py + F[2];
    float bb = F[3]*px + F[4]*py + F[5];
    float cc = F[6]*px + F[7]*py + F[8];
    const float eps = 1e-8f;
    float y0 = -cc / (bb + eps);
    float yW = -(aa*(WW - 1) + cc) / (bb + eps);
    float x0 = -cc / (aa + eps);
    float xH = -(bb*(HH - 1) + cc) / (aa + eps);
    bool vv[4];
    vv[0] = (y0 >= 0.f) && (y0 < (float)HH);
    vv[1] = (yW >= 0.f) && (yW < (float)HH);
    vv[2] = (x0 >= 0.f) && (x0 < (float)WW);
    vv[3] = (xH >= 0.f) && (xH < (float)WW);
    float ptx[4] = { 0.f, (float)(WW - 1), x0, xH };
    float pty[4] = { y0, yW, 0.f, (float)(HH - 1) };
    int i1 = 0, i2 = 0, cnt = 0;
    #pragma unroll
    for (int i = 0; i < 4; ++i)
      if (vv[i]) { if (cnt == 0) i1 = i; else if (cnt == 1) i2 = i; ++cnt; }
    float cx = (WW - 1) * 0.5f, cy = (HH - 1) * 0.5f;
    float p1x, p1y, p2x, p2y;
    if (cnt < 2) { p1x = p2x = cx; p1y = p2y = cy; }
    else { p1x = ptx[i1]; p1y = pty[i1]; p2x = ptx[i2]; p2y = pty[i2]; }
    float s = (float)n / (float)(NS - 1);
    sx[t] = p1x + (p2x - p1x) * s;
    sy[t] = p1y + (p2y - p1y) * s;
  }
  __syncthreads();

  // single-channel bilinear tap (zero outside, clipped index)
#define TAP1(base, ch, dx, dy, wt, o0)                                     \
  { int xx = xi + (dx), yy = yi + (dy);                                    \
    bool val = (xx >= 0) && (xx <= WW - 1) && (yy >= 0) && (yy <= HH - 1); \
    int xc = min(max(xx, 0), WW - 1), yc = min(max(yy, 0), HH - 1);        \
    const float* p = (base) + (size_t)(yc * WW + xc) * CQKV;               \
    float m = val ? (wt) : 0.0f;                                           \
    (o0) += m * p[(ch)]; }

  // ---- stage 2: logits, wave `h` handles head h; lanes split DK in pairs ----
  {
    int h = wave;
    const float* qp = QKV + (size_t)blk * CQKV + h * DK;
    int c0 = 2 * lane, c1 = 2 * lane + 1;
    float q0 = qp[c0], q1 = qp[c1];
    const float scale = 0.125f; // 1/sqrt(64)
    for (int n = 0; n < NTOT; ++n) {
      int j   = n / NS;
      int src = j + (j >= ref ? 1 : 0);
      const float* Kb = QKV + (size_t)(b*SS + src) * HW * CQKV + 512 + h * DK;
      float X = sx[n], Y = sy[n];
      float xf = floorf(X), yf = floorf(Y);
      float fx = X - xf, fy = Y - yf;
      int xi = (int)xf, yi = (int)yf;
      float w00 = (1.f - fx) * (1.f - fy), w10 = fx * (1.f - fy);
      float w01 = (1.f - fx) * fy,         w11 = fx * fy;
      float k0 = 0.f, k1 = 0.f;
      TAP1(Kb, c0, 0, 0, w00, k0) TAP1(Kb, c1, 0, 0, w00, k1)
      TAP1(Kb, c0, 1, 0, w10, k0) TAP1(Kb, c1, 1, 0, w10, k1)
      TAP1(Kb, c0, 0, 1, w01, k0) TAP1(Kb, c1, 0, 1, w01, k1)
      TAP1(Kb, c0, 1, 1, w11, k0) TAP1(Kb, c1, 1, 1, w11, k1)
      float acc = q0 * k0 + q1 * k1;
      #pragma unroll
      for (int m = 16; m >= 1; m >>= 1) acc += __shfl_xor(acc, m, 32);
      if (lane == 0) slog[h * NTOT + n] = acc * scale;
    }
  }
  __syncthreads();

  // ---- stage 3: softmax over 96 samples per head (f32) ----
  {
    int h = wave;
    float mx = -1e30f;
    for (int n = lane; n < NTOT; n += 32) mx = fmaxf(mx, slog[h * NTOT + n]);
    #pragma unroll
    for (int m = 16; m >= 1; m >>= 1) mx = fmaxf(mx, __shfl_xor(mx, m, 32));
    float sum = 0.f;
    for (int n = lane; n < NTOT; n += 32) {
      float e = __expf(slog[h * NTOT + n] - mx);
      slog[h * NTOT + n] = e;
      sum += e;
    }
    #pragma unroll
    for (int m = 16; m >= 1; m >>= 1) sum += __shfl_xor(sum, m, 32);
    float inv = 1.0f / sum;
    for (int n = lane; n < NTOT; n += 32) slog[h * NTOT + n] *= inv;
  }
  __syncthreads();

  // ---- stage 4: weighted V accumulation; thread = (head, dv channel) ----
  {
    int h = wave;
    float acc = 0.f;
    for (int n = 0; n < NTOT; ++n) {
      int j   = n / NS;
      int src = j + (j >= ref ? 1 : 0);
      const float* Vb = QKV + (size_t)(b*SS + src) * HW * CQKV + 1024 + h * DV;
      float wgt = slog[h * NTOT + n];
      float X = sx[n], Y = sy[n];
      float xf = floorf(X), yf = floorf(Y);
      float fx = X - xf, fy = Y - yf;
      int xi = (int)xf, yi = (int)yf;
      float w00 = (1.f - fx) * (1.f - fy), w10 = fx * (1.f - fy);
      float w01 = (1.f - fx) * fy,         w11 = fx * fy;
      float vs = 0.f;
      TAP1(Vb, lane, 0, 0, w00, vs)
      TAP1(Vb, lane, 1, 0, w10, vs)
      TAP1(Vb, lane, 0, 1, w01, vs)
      TAP1(Vb, lane, 1, 1, w11, vs)
      acc += wgt * vs;
    }
    obf[(size_t)blk * (NHEAD * DV) + h * DV + lane] = (__bf16)acc;
  }
#undef TAP1
}

// ---------------- launcher ----------------
extern "C" void kernel_launch(void* const* d_in, const int* in_sizes, int n_in,
                              void* d_out, int out_size, void* d_ws, size_t ws_size,
                              hipStream_t stream) {
  (void)in_sizes; (void)n_in; (void)out_size; (void)ws_size;
  const float* x     = (const float*)d_in[0];
  const float* intr  = (const float*)d_in[1];
  const float* ext   = (const float*)d_in[2];
  const float* q_w   = (const float*)d_in[3];
  const float* q_b   = (const float*)d_in[4];
  const float* k_w   = (const float*)d_in[5];
  const float* k_b   = (const float*)d_in[6];
  const float* v_w   = (const float*)d_in[7];
  const float* v_b   = (const float*)d_in[8];
  const float* out_w = (const float*)d_in[9];
  const float* out_b = (const float*)d_in[10];
  const float* zc_w  = (const float*)d_in[11];
  const float* zc_b  = (const float*)d_in[12];
  float* outp = (float*)d_out;

  // workspace carve-up (256B aligned chunks)
  char* ws = (char*)d_ws;
  size_t off = 0;
  auto carve = [&](size_t bytes) -> void* {
    void* p = ws + off;
    off += (bytes + 255) & ~(size_t)255;
    return p;
  };
  __bf16* xbf   = (__bf16*)carve((size_t)NROW * CIN * 2);
  __bf16* wqkv  = (__bf16*)carve((size_t)CQKV * CIN * 2);
  float*  bqkv  = (float*) carve((size_t)CQKV * 4);
  float*  QKV   = (float*) carve((size_t)NROW * CQKV * 4);
  float*  Fm    = (float*) carve((size_t)BB * SS * SS * 9 * 4);
  __bf16* obf   = (__bf16*)carve((size_t)NROW * (NHEAD * DV) * 2);
  __bf16* w1    = (__bf16*)carve((size_t)OC * (NHEAD * DV) * 2);
  __bf16* w2    = (__bf16*)carve((size_t)OC * OC * 2);
  __bf16* tmpbf = (__bf16*)carve((size_t)NROW * OC * 2);

  auto cvt = [&](const float* s, __bf16* d, int n) {
    f32_to_bf16_kernel<<<(n + 255) / 256, 256, 0, stream>>>(s, d, n);
  };
  // bf16 mirrors of x and all GEMM weights (wqkv = [q_w; k_w; v_w])
  cvt(x,     xbf,                 NROW * CIN);
  cvt(q_w,   wqkv,                512 * CIN);
  cvt(k_w,   wqkv + 512 * CIN,    512 * CIN);
  cvt(v_w,   wqkv + 1024 * CIN,   256 * CIN);
  cvt(out_w, w1,                  OC * (NHEAD * DV));
  cvt(zc_w,  w2,                  OC * OC);
  // concat biases
  f32_copy_kernel<<<2, 256, 0, stream>>>(q_b, bqkv,        512);
  f32_copy_kernel<<<2, 256, 0, stream>>>(k_b, bqkv + 512,  512);
  f32_copy_kernel<<<1, 256, 0, stream>>>(v_b, bqkv + 1024, 256);

  // fundamental matrices
  fmat_kernel<<<1, 32, 0, stream>>>(intr, ext, Fm);

  // QKV projection: (4608 x 512) * (1280 x 512)^T  -> f32, channel-last
  {
    dim3 grid(CQKV / 32, NROW / 128);
    gemm_bf16_kernel<<<grid, 256, 0, stream>>>(xbf, wqkv, bqkv, QKV, nullptr,
                                               NROW, CQKV, CIN);
  }

  // epipolar attention (one block per (b, ref, pixel))
  attn_kernel<<<NROW, 256, 0, stream>>>(QKV, Fm, obf);

  // out projection 1: (4608 x 256) * (512 x 256)^T -> bf16 intermediate
  {
    dim3 grid(OC / 32, NROW / 128);
    gemm_bf16_kernel<<<grid, 256, 0, stream>>>(obf, w1, out_b, nullptr, tmpbf,
                                               NROW, OC, NHEAD * DV);
  }
  // out projection 2: (4608 x 512) * (512 x 512)^T -> f32 final output
  {
    dim3 grid(OC / 32, NROW / 128);
    gemm_bf16_kernel<<<grid, 256, 0, stream>>>(tmpbf, w2, zc_b, outp, nullptr,
                                               NROW, OC, OC);
  }
}
